// GCNWithRegularization_54915451846875
// MI455X (gfx1250) — matrix operand order repaired and checked by reading, not compiled, not run on hardware
//
#include <hip/hip_runtime.h>
#include <math.h>

typedef __attribute__((ext_vector_type(16))) __bf16 v16bf;
typedef __attribute__((ext_vector_type(8)))  float  v8f;

// ---------------------------------------------------------------------------
// WMMA GEMM: C[M,Ncols] = A[M,K] @ B[K,Ncols], f32 row-major in memory.
// Split-precision bf16: each f32 operand = hi + lo (bf16 pair); accumulate
// hi*hi + hi*lo + lo*hi with v_wmma_f32_16x16x32_bf16 (f32 accumulators)
// => ~fp32 accuracy. One wave32 per 16x16 C tile. K % 32 == 0.
// Out-of-range rows/cols are clamped for loads (their results only land in
// lanes/rows that the guarded store discards), so the K-loop is branch-free.
// ---------------------------------------------------------------------------
__global__ __launch_bounds__(32) void gemm_wmma_bf16x3(
    const float* __restrict__ A, const float* __restrict__ B,
    float* __restrict__ C, int M, int Ncols, int K)
{
  const int tileM = blockIdx.x * 16;
  const int tileN = blockIdx.y * 16;
  const int lane  = threadIdx.x;        // 0..31
  const int half  = lane >> 4;          // 0: lanes 0-15, 1: lanes 16-31
  const int l     = lane & 15;

  const int row  = tileM + l;           // logical A row for this lane
  const int coln = tileN + l;           // logical B/C column for this lane
  const int rowc = (row  < M)     ? row  : (M - 1);      // clamped (loads only)
  const int colc = (coln < Ncols) ? coln : (Ncols - 1);  // clamped (loads only)

  v8f acc = {};
  const float* arow0 = A + (size_t)rowc * K + half * 16;
  const float* bcol0 = B + (size_t)(half * 16) * Ncols + colc;

  for (int k0 = 0; k0 < K; k0 += 32) {
    // ---- A fragment (16x32 bf16): lanes 0-15 hold K=k0+0..15;
    //      lanes 16-31 hold K=k0+16..31 (row = l).
    v16bf a_hi, a_lo;
    const float* arow = arow0 + k0;
#pragma unroll
    for (int i = 0; i < 8; ++i) {
      float v0 = arow[i];
      float v1 = arow[8 + i];
      float h0 = (float)(__bf16)v0;
      float h1 = (float)(__bf16)v1;
      a_hi[i]     = (__bf16)v0;        a_lo[i]     = (__bf16)(v0 - h0);
      a_hi[8 + i] = (__bf16)v1;        a_lo[8 + i] = (__bf16)(v1 - h1);
    }
    // ---- B fragment (32x16 bf16): VGPR j holds K=k0+2j,2j+1 (lanes 0-15)
    //      or K=k0+16+2j,17+2j (lanes 16-31), col = l.
    v16bf b_hi, b_lo;
    const float* bcol = bcol0 + (size_t)k0 * Ncols;
#pragma unroll
    for (int j = 0; j < 16; ++j) {
      float v = bcol[(size_t)j * Ncols];
      float h = (float)(__bf16)v;
      b_hi[j] = (__bf16)v;
      b_lo[j] = (__bf16)(v - h);
    }
    acc = __builtin_amdgcn_wmma_f32_16x16x32_bf16(
        false, a_hi, false, b_hi, (short)0, acc, false, false);
    acc = __builtin_amdgcn_wmma_f32_16x16x32_bf16(
        false, a_hi, false, b_lo, (short)0, acc, false, false);
    acc = __builtin_amdgcn_wmma_f32_16x16x32_bf16(
        false, a_lo, false, b_hi, (short)0, acc, false, false);
  }

  // ---- C 16x16 f32: VGPR r -> row tileM + r + half*8, col = tileN + l
  if (coln < Ncols) {
#pragma unroll
    for (int r = 0; r < 8; ++r) {
      int orow = tileM + r + half * 8;
      if (orow < M) C[(size_t)orow * Ncols + coln] = acc[r];
    }
  }
}

// ---------------------------------------------------------------------------
// Degree counting (in-degree for norm + CSR-by-dst, out-degree for CSR-by-src)
// ---------------------------------------------------------------------------
__global__ void count_deg(const int* __restrict__ src, const int* __restrict__ dst,
                          int E, unsigned* __restrict__ deg_dst,
                          unsigned* __restrict__ deg_src)
{
  int e = blockIdx.x * blockDim.x + threadIdx.x;
  if (e < E) {
    atomicAdd(&deg_dst[dst[e]], 1u);
    atomicAdd(&deg_src[src[e]], 1u);
  }
}

// Single-block exclusive scan over n entries -> out[0..n] (out[n] = total)
__global__ __launch_bounds__(1024) void exscan(const unsigned* __restrict__ in,
                                               int n, unsigned* __restrict__ out)
{
  __shared__ unsigned part[1024];
  int t = threadIdx.x;
  int chunk = (n + 1023) / 1024;
  int b0 = t * chunk;
  int b1 = b0 + chunk; if (b1 > n) b1 = n;
  unsigned s = 0;
  for (int i = b0; i < b1; ++i) s += in[i];
  part[t] = s;
  __syncthreads();
  for (int off = 1; off < 1024; off <<= 1) {
    unsigned v = (t >= off) ? part[t - off] : 0u;
    __syncthreads();
    part[t] += v;
    __syncthreads();
  }
  unsigned run = part[t] - s;           // exclusive base
  for (int i = b0; i < b1; ++i) { out[i] = run; run += in[i]; }
  if (t == 1023) out[n] = part[1023];
}

__global__ void compute_dinv(const unsigned* __restrict__ deg_dst,
                             float* __restrict__ dinv, int n)
{
  int v = blockIdx.x * blockDim.x + threadIdx.x;
  if (v < n) dinv[v] = rsqrtf((float)(deg_dst[v] + 1u));  // +1 self-loop
}

__global__ void fill_csr(const int* __restrict__ src, const int* __restrict__ dst,
                         int E,
                         const unsigned* __restrict__ rp_dst,
                         const unsigned* __restrict__ rp_src,
                         unsigned* __restrict__ cur_dst,
                         unsigned* __restrict__ cur_src,
                         int* __restrict__ col_dst, int* __restrict__ col_src)
{
  int e = blockIdx.x * blockDim.x + threadIdx.x;
  if (e < E) {
    int s = src[e], d = dst[e];
    unsigned p = rp_dst[d] + atomicAdd(&cur_dst[d], 1u);
    col_dst[p] = s;
    unsigned q = rp_src[s] + atomicAdd(&cur_src[s], 1u);
    col_src[q] = d;
  }
}

// ---------------------------------------------------------------------------
// Layer-1 aggregation: out[v,f] = relu( bias[f] + dinv[v]^2 * h[v,f]
//                                       + sum_e dinv[s]*dinv[v]*h[s,f] )
// Block = node, thread = feature (F = 128). All gathers hit L2 (h fits).
// ---------------------------------------------------------------------------
__global__ __launch_bounds__(128) void agg_relu_bias(
    const float* __restrict__ hin, const unsigned* __restrict__ rowptr,
    const int* __restrict__ col, const float* __restrict__ dinv,
    const float* __restrict__ bias, float* __restrict__ hout, int F)
{
  int v = blockIdx.x;
  int f = threadIdx.x;
  float dv  = dinv[v];
  float acc = hin[(size_t)v * F + f] * dv * dv;     // self loop
  unsigned beg = rowptr[v], end = rowptr[v + 1];
  for (unsigned e = beg; e < end; ++e) {
    int s = col[e];
    acc += hin[(size_t)s * F + f] * (dinv[s] * dv);
  }
  acc += bias[f];
  hout[(size_t)v * F + f] = fmaxf(acc, 0.0f);
}

// ---------------------------------------------------------------------------
// Layer-2 aggregation + bias + log_softmax fused (F = 40, block = 64 threads)
// ---------------------------------------------------------------------------
__global__ __launch_bounds__(64) void agg2_logsoftmax(
    const float* __restrict__ hin, const unsigned* __restrict__ rowptr,
    const int* __restrict__ col, const float* __restrict__ dinv,
    const float* __restrict__ bias, float* __restrict__ out, int F)
{
  __shared__ float logit[64];
  int v = blockIdx.x;
  int f = threadIdx.x;
  if (f < F) {
    float dv  = dinv[v];
    float acc = hin[(size_t)v * F + f] * dv * dv;
    unsigned beg = rowptr[v], end = rowptr[v + 1];
    for (unsigned e = beg; e < end; ++e) {
      int s = col[e];
      acc += hin[(size_t)s * F + f] * (dinv[s] * dv);
    }
    logit[f] = acc + bias[f];
  }
  __syncthreads();
  if (f < F) {
    float m = -3.402823466e38f;
    for (int i = 0; i < F; ++i) m = fmaxf(m, logit[i]);
    float s = 0.0f;
    for (int i = 0; i < F; ++i) s += __expf(logit[i] - m);
    out[(size_t)v * F + f] = logit[f] - m - __logf(s);
  }
}

// ---------------------------------------------------------------------------
// reg = sum_{i,j} A[i,j]*A[j,i]: per edge instance (s,d), count instances (d,s)
// via CSR-by-src row of d. Integer accumulate -> exact.
// ---------------------------------------------------------------------------
__global__ void reg_count(const int* __restrict__ src, const int* __restrict__ dst,
                          int E, const unsigned* __restrict__ rp_src,
                          const int* __restrict__ col_src, unsigned* __restrict__ reg)
{
  int e = blockIdx.x * blockDim.x + threadIdx.x;
  unsigned c = 0;
  if (e < E) {
    int s = src[e], d = dst[e];
    unsigned beg = rp_src[d], end = rp_src[d + 1];
    for (unsigned i = beg; i < end; ++i)
      if (col_src[i] == s) ++c;
  }
  if (c) atomicAdd(reg, c);
}

__global__ void finalize_reg(const unsigned* __restrict__ reg,
                             float* __restrict__ out_scalar)
{
  if (threadIdx.x == 0 && blockIdx.x == 0) *out_scalar = (float)(*reg);
}

// ---------------------------------------------------------------------------
extern "C" void kernel_launch(void* const* d_in, const int* in_sizes, int n_in,
                              void* d_out, int out_size, void* d_ws, size_t ws_size,
                              hipStream_t stream)
{
  const int FIN = 512, FH = 128, FOUT = 40;
  const int N = in_sizes[0] / FIN;       // 10000
  const int E = in_sizes[1] / 2;         // 320000

  const float* x   = (const float*)d_in[0];
  const int*   ei  = (const int*)d_in[1];
  const int*   src = ei;
  const int*   dst = ei + E;
  const float* W1  = (const float*)d_in[2];
  const float* b1  = (const float*)d_in[3];
  const float* W2  = (const float*)d_in[4];
  const float* b2  = (const float*)d_in[5];
  float* out = (float*)d_out;            // [N*FOUT] log_softmax ++ [1] reg

  // -------- workspace carve-up --------
  char* p = (char*)d_ws;
  float*    h_lin1  = (float*)p;    p += (size_t)N * FH * 4;
  float*    h_agg1  = (float*)p;    p += (size_t)N * FH * 4;
  float*    h_lin2  = (float*)p;    p += (size_t)N * FOUT * 4;
  unsigned* deg_dst = (unsigned*)p; p += (size_t)N * 4;
  unsigned* deg_src = (unsigned*)p; p += (size_t)N * 4;
  unsigned* rp_dst  = (unsigned*)p; p += (size_t)(N + 1) * 4;
  unsigned* rp_src  = (unsigned*)p; p += (size_t)(N + 1) * 4;
  unsigned* cur_dst = (unsigned*)p; p += (size_t)N * 4;
  unsigned* cur_src = (unsigned*)p; p += (size_t)N * 4;
  float*    dinv    = (float*)p;    p += (size_t)N * 4;
  int*      col_dst = (int*)p;      p += (size_t)E * 4;
  int*      col_src = (int*)p;      p += (size_t)E * 4;
  unsigned* reg     = (unsigned*)p; p += 4;

  // -------- zero the counters (graph-capture-safe) --------
  hipMemsetAsync(deg_dst, 0, (size_t)N * 4, stream);
  hipMemsetAsync(deg_src, 0, (size_t)N * 4, stream);
  hipMemsetAsync(cur_dst, 0, (size_t)N * 4, stream);
  hipMemsetAsync(cur_src, 0, (size_t)N * 4, stream);
  hipMemsetAsync(reg,     0, 4, stream);

  const int TB = 256;
  const int egrid = (E + TB - 1) / TB;
  const int ngrid = (N + TB - 1) / TB;

  // -------- graph structure --------
  count_deg<<<egrid, TB, 0, stream>>>(src, dst, E, deg_dst, deg_src);
  exscan<<<1, 1024, 0, stream>>>(deg_dst, N, rp_dst);
  exscan<<<1, 1024, 0, stream>>>(deg_src, N, rp_src);
  compute_dinv<<<ngrid, TB, 0, stream>>>(deg_dst, dinv, N);
  fill_csr<<<egrid, TB, 0, stream>>>(src, dst, E, rp_dst, rp_src,
                                     cur_dst, cur_src, col_dst, col_src);

  // -------- layer 1: X @ W1 (WMMA bf16x3), aggregate, +b1, relu --------
  {
    dim3 grid((N + 15) / 16, FH / 16);
    gemm_wmma_bf16x3<<<grid, 32, 0, stream>>>(x, W1, h_lin1, N, FH, FIN);
  }
  agg_relu_bias<<<N, FH, 0, stream>>>(h_lin1, rp_dst, col_dst, dinv, b1,
                                      h_agg1, FH);

  // -------- layer 2: H @ W2 (WMMA bf16x3), aggregate, +b2, log_softmax ----
  {
    dim3 grid((N + 15) / 16, (FOUT + 15) / 16);
    gemm_wmma_bf16x3<<<grid, 32, 0, stream>>>(h_agg1, W2, h_lin2, N, FOUT, FH);
  }
  agg2_logsoftmax<<<N, 64, 0, stream>>>(h_lin2, rp_dst, col_dst, dinv, b2,
                                        out, FOUT);

  // -------- regularizer --------
  reg_count<<<egrid, TB, 0, stream>>>(src, dst, E, rp_src, col_src, reg);
  finalize_reg<<<1, 32, 0, stream>>>(reg, out + (size_t)N * FOUT);
}
